// SwinTransformerBlock_1846835937296
// MI455X (gfx1250) — compile-verified, matrix operand order checked by reading, not run on hardware
//
#include <hip/hip_runtime.h>
#include <math.h>

// ---------------- constants ----------------
#define DIM        384
#define HEADS      12
#define HEAD_DIM   32
#define WS         7
#define SHIFT      3
#define NTOK       49            // tokens per window
#define NPAD       64            // padded tokens per window
#define MLP_HIDDEN 1536
#define BATCH      32
#define HH         56
#define WW         56
#define L_TOK      (HH*WW)       // 3136
#define M_TOK      (BATCH*L_TOK) // 100352 rows, divisible by 128
#define NWIN       64            // 8x8 windows per image
#define NCLS       4             // border-row x border-col window classes
#define SCALE_Q    0.17677669529663687f   // 32^-0.5

typedef __attribute__((ext_vector_type(16))) __bf16 v16bf;
typedef __attribute__((ext_vector_type(16))) short  v16s;
typedef __attribute__((ext_vector_type(8)))  float  v8f;
typedef __attribute__((ext_vector_type(8)))  int    v8i;

__device__ __forceinline__ unsigned short f32_to_bf16(float f) {
  union { float f; unsigned u; } v; v.f = f;
  unsigned r = v.u + 0x7FFFu + ((v.u >> 16) & 1u);
  return (unsigned short)(r >> 16);
}

__device__ __forceinline__ v8f zero8() { v8f z = {0.f,0.f,0.f,0.f,0.f,0.f,0.f,0.f}; return z; }

__device__ __forceinline__ v8f wmma_bf16(v16bf a, v16bf b, v8f c) {
  return __builtin_amdgcn_wmma_f32_16x16x32_bf16(false, a, false, b, (short)0, c, false, false);
}

// A/B fragment for 16x16x32 bf16 WMMA from a row-major [rows][ldk] bf16 matrix.
// ISA 16-bit A layout: lane L holds row (L&15); lanes>=16 offset k by 8;
// vgpr0-3 -> k base..base+7, vgpr4-7 -> k 16+base..16+base+7 (contiguous pairs).
__device__ __forceinline__ v16bf load_frag_rowmajor(const unsigned short* base,
                                                    int row, int ldk, int koff, int lane) {
  int r  = row + (lane & 15);
  int hb = (lane & 16) ? 8 : 0;
  const unsigned short* p = base + (size_t)r * ldk + koff + hb;
  v8i t;
#pragma unroll
  for (int j = 0; j < 4; ++j) t[j]     = *(const unsigned int*)(p + 2*j);
#pragma unroll
  for (int j = 0; j < 4; ++j) t[4 + j] = *(const unsigned int*)(p + 16 + 2*j);
  return __builtin_bit_cast(v16bf, t);
}

// B fragment where B is stored row-major [K][ld] and lane indexes the N column
// (strided gather; used for P@V where V is token-major in LDS).
__device__ __forceinline__ v16bf load_frag_colstrided(const unsigned short* base,
                                                      int col, int ld, int koff, int lane) {
  int n  = col + (lane & 15);
  int hb = (lane & 16) ? 8 : 0;
  v16s t;
#pragma unroll
  for (int i = 0; i < 16; ++i) {
    int k = koff + ((i < 8) ? (hb + i) : (16 + hb + (i - 8)));
    t[i] = (short)base[(size_t)k * ld + n];
  }
  return __builtin_bit_cast(v16bf, t);
}

// ---------------- weight convert with transpose: [K][N] f32 -> [N][K] bf16 ----------------
__global__ void cvt_transpose_kernel(const float* __restrict__ in,
                                     unsigned short* __restrict__ out, int K, int N) {
  int i = blockIdx.x * blockDim.x + threadIdx.x;
  if (i < K * N) {
    int n = i / K, k = i % K;
    out[i] = f32_to_bf16(in[(size_t)k * N + n]);
  }
}

// ---------------- fused rel-pos-bias + shift-mask + key-pad table ----------------
// table[cls][head][n][m]; cls = 2*(window is last row) + (window is last col)
__global__ void build_biasmask_kernel(const float* __restrict__ rel_table,
                                      float* __restrict__ table) {
  int i = blockIdx.x * blockDim.x + threadIdx.x;
  if (i >= NCLS * HEADS * NPAD * NPAD) return;
  int m    = i & 63;
  int n    = (i >> 6) & 63;
  int head = (i >> 12) % HEADS;
  int cls  = (i >> 12) / HEADS;
  float v;
  if (n >= NTOK) {
    v = 0.0f;                       // padded query rows: keep finite
  } else if (m >= NTOK) {
    v = -1e30f;                     // padded key cols: exp -> 0
  } else {
    int rn = n / WS, cn = n % WS, rm = m / WS, cm = m % WS;
    v = rel_table[((rn - rm + WS - 1) * (2 * WS - 1) + (cn - cm + WS - 1)) * HEADS + head];
    int bh = cls >> 1, bw = cls & 1;
    // region ids on shifted grid; interior windows (wh<7 / ww<7) are wholly region 0
    int gn = (bh ? ((rn < WS - SHIFT) ? 1 : 2) : 0) * 3 + (bw ? ((cn < WS - SHIFT) ? 1 : 2) : 0);
    int gm = (bh ? ((rm < WS - SHIFT) ? 1 : 2) : 0) * 3 + (bw ? ((cm < WS - SHIFT) ? 1 : 2) : 0);
    if (gn != gm) v -= 100.0f;
  }
  table[i] = v;
}

// ---------------- LayerNorm1 (token per block) ----------------
__global__ __launch_bounds__(128) void ln1_kernel(const float* __restrict__ x,
                                                  const float* __restrict__ g,
                                                  const float* __restrict__ b,
                                                  unsigned short* __restrict__ out) {
  __shared__ float red[128];
  int t = blockIdx.x, tid = threadIdx.x;
  const float* xr = x + (size_t)t * DIM;
  float v0 = xr[tid], v1 = xr[tid + 128], v2 = xr[tid + 256];
  red[tid] = v0 + v1 + v2;
  __syncthreads();
  for (int off = 64; off; off >>= 1) { if (tid < off) red[tid] += red[tid + off]; __syncthreads(); }
  float mean = red[0] * (1.0f / DIM);
  __syncthreads();
  float d0 = v0 - mean, d1 = v1 - mean, d2 = v2 - mean;
  red[tid] = d0*d0 + d1*d1 + d2*d2;
  __syncthreads();
  for (int off = 64; off; off >>= 1) { if (tid < off) red[tid] += red[tid + off]; __syncthreads(); }
  float inv = rsqrtf(red[0] * (1.0f / DIM) + 1e-5f);
  unsigned short* o = out + (size_t)t * DIM;
  o[tid]       = f32_to_bf16(d0 * inv * g[tid]       + b[tid]);
  o[tid + 128] = f32_to_bf16(d1 * inv * g[tid + 128] + b[tid + 128]);
  o[tid + 256] = f32_to_bf16(d2 * inv * g[tid + 256] + b[tid + 256]);
}

// ---------------- window-reverse + residual + LayerNorm2 ----------------
__global__ __launch_bounds__(128) void residual_ln2_kernel(const float* __restrict__ x,
                                                           const float* __restrict__ projout, // windowed rows
                                                           const float* __restrict__ g,
                                                           const float* __restrict__ b,
                                                           float* __restrict__ hbuf,
                                                           unsigned short* __restrict__ ln2out) {
  __shared__ float red[128];
  int t = blockIdx.x, tid = threadIdx.x;
  int bb  = t / L_TOK, rem = t % L_TOK;
  int hh  = rem / WW,  ww  = rem % WW;
  int hs  = (hh + HH - SHIFT) % HH;       // inverse of final roll(+SHIFT)
  int wsv = (ww + WW - SHIFT) % WW;
  size_t winrow = (((size_t)bb * 8 + hs / WS) * 8 + wsv / WS) * NTOK + (hs % WS) * WS + (wsv % WS);
  const float* xr = x + (size_t)t * DIM;
  const float* pr = projout + winrow * DIM;
  float v0 = xr[tid]       + pr[tid];
  float v1 = xr[tid + 128] + pr[tid + 128];
  float v2 = xr[tid + 256] + pr[tid + 256];
  float* hr = hbuf + (size_t)t * DIM;
  hr[tid] = v0; hr[tid + 128] = v1; hr[tid + 256] = v2;
  red[tid] = v0 + v1 + v2;
  __syncthreads();
  for (int off = 64; off; off >>= 1) { if (tid < off) red[tid] += red[tid + off]; __syncthreads(); }
  float mean = red[0] * (1.0f / DIM);
  __syncthreads();
  float d0 = v0 - mean, d1 = v1 - mean, d2 = v2 - mean;
  red[tid] = d0*d0 + d1*d1 + d2*d2;
  __syncthreads();
  for (int off = 64; off; off >>= 1) { if (tid < off) red[tid] += red[tid + off]; __syncthreads(); }
  float inv = rsqrtf(red[0] * (1.0f / DIM) + 1e-5f);
  unsigned short* o = ln2out + (size_t)t * DIM;
  o[tid]       = f32_to_bf16(d0 * inv * g[tid]       + b[tid]);
  o[tid + 128] = f32_to_bf16(d1 * inv * g[tid + 128] + b[tid + 128]);
  o[tid + 256] = f32_to_bf16(d2 * inv * g[tid + 256] + b[tid + 256]);
}

// ---------------- tiled bf16 WMMA GEMM, 128x128 tile, 8 waves ----------------
// B is pre-transposed (N-major), so both tiles stage with b128 loads.
// MODE 0: qkv   -> bf16 out, bias, cols<384 scaled by SCALE_Q
// MODE 1: proj  -> f32 out, bias
// MODE 2: fc1   -> bf16 out, bias + exact GELU
// MODE 3: fc2   -> f32 out, bias + residual (hbuf)
template <int MODE>
__global__ __launch_bounds__(256) void gemm_wmma(const unsigned short* __restrict__ A,   // M x K
                                                 const unsigned short* __restrict__ BT,  // N x K
                                                 const float* __restrict__ bias,
                                                 float* __restrict__ outF,
                                                 unsigned short* __restrict__ outB,
                                                 const float* __restrict__ resid,
                                                 int M, int K, int N) {
  __shared__ unsigned short ldsA[128 * 32];
  __shared__ unsigned short ldsBt[128 * 32];
  const int bm = blockIdx.y * 128;
  const int bn = blockIdx.x * 128;
  const int tid = threadIdx.x, lane = tid & 31, wave = tid >> 5;
  const int wm = (wave & 3) * 32;   // wave row offset inside tile
  const int wn = (wave >> 2) * 64;  // wave col offset inside tile

  const int row  = tid >> 1, half = (tid & 1) * 16;
  const unsigned short* srcA = A  + (size_t)(bm + row) * K + half;
  const unsigned short* srcB = BT + (size_t)(bn + row) * K + half;

  v8f acc[2][4];
#pragma unroll
  for (int mt = 0; mt < 2; ++mt)
#pragma unroll
    for (int nt = 0; nt < 4; ++nt) acc[mt][nt] = zero8();

  for (int k0 = 0; k0 < K; k0 += 32) {
    __syncthreads();
    *(uint4*)&ldsA [row * 32 + half]     = *(const uint4*)(srcA + k0);
    *(uint4*)&ldsA [row * 32 + half + 8] = *(const uint4*)(srcA + k0 + 8);
    *(uint4*)&ldsBt[row * 32 + half]     = *(const uint4*)(srcB + k0);
    *(uint4*)&ldsBt[row * 32 + half + 8] = *(const uint4*)(srcB + k0 + 8);
    if (k0 + 32 < K) {                     // prefetch next k-step tiles
      __builtin_prefetch(srcA + k0 + 32, 0, 0);
      __builtin_prefetch(srcB + k0 + 32, 0, 0);
    }
    __syncthreads();

    v16bf bfrag[4];
#pragma unroll
    for (int nt = 0; nt < 4; ++nt)
      bfrag[nt] = load_frag_rowmajor(ldsBt, wn + nt * 16, 32, 0, lane);
#pragma unroll
    for (int mt = 0; mt < 2; ++mt) {
      v16bf a = load_frag_rowmajor(ldsA, wm + mt * 16, 32, 0, lane);
#pragma unroll
      for (int nt = 0; nt < 4; ++nt)
        acc[mt][nt] = wmma_bf16(a, bfrag[nt], acc[mt][nt]);
    }
  }

  // epilogue (C/D layout: n = lane&15, m = r + 8*(lane>=16))
#pragma unroll
  for (int mt = 0; mt < 2; ++mt)
#pragma unroll
    for (int nt = 0; nt < 4; ++nt)
#pragma unroll
      for (int r = 0; r < 8; ++r) {
        int rrow = bm + wm + mt * 16 + r + ((lane >> 4) << 3);
        int col  = bn + wn + nt * 16 + (lane & 15);
        float v = acc[mt][nt][r] + bias[col];
        size_t idx = (size_t)rrow * N + col;
        if (MODE == 0) {
          if (col < DIM) v *= SCALE_Q;        // q columns
          outB[idx] = f32_to_bf16(v);
        } else if (MODE == 1) {
          outF[idx] = v;
        } else if (MODE == 2) {
          v = 0.5f * v * (1.0f + erff(v * 0.70710678118654752f));
          outB[idx] = f32_to_bf16(v);
        } else {
          outF[idx] = v + resid[idx];
        }
      }
}

// ---------------- windowed attention: one wave per (window, head) ----------------
// LDS arena (28KB): Q[4K] K[4K] V[4K] S[16K]; P(8K) overlays dead Q+K after S phase.
__global__ __launch_bounds__(32) void attn_kernel(const unsigned short* __restrict__ qkv, // [M_TOK][1152] bf16
                                                  const float* __restrict__ bm_table,      // [4][12][64][64]
                                                  unsigned short* __restrict__ attnout) {  // [2048*49][384] bf16
  __shared__ __align__(16) char arena[28 * 1024];
  unsigned short* ldsQ = (unsigned short*)arena;
  unsigned short* ldsK = (unsigned short*)(arena + 4096);
  unsigned short* ldsV = (unsigned short*)(arena + 8192);
  float*          ldsS = (float*)(arena + 12288);
  unsigned short* ldsP = (unsigned short*)arena;   // overlays Q,K

  const int lane = threadIdx.x;
  const int bid  = blockIdx.x;
  const int win  = bid / HEADS;
  const int head = bid % HEADS;
  const int bb   = win / NWIN;
  const int wIdx = win % NWIN;
  const int wh   = wIdx / 8, wwin = wIdx % 8;
  const int cls  = ((wh == 7) ? 2 : 0) + ((wwin == 7) ? 1 : 0);
  const float* tbase = bm_table + (((size_t)cls * HEADS + head) << 12);

  // gather shifted/rolled q,k,v rows for this window+head into LDS (zero-pad 49..63)
  for (int r = lane; r < NPAD; r += 32) {
    unsigned int* q32 = (unsigned int*)&ldsQ[r * HEAD_DIM];
    unsigned int* k32 = (unsigned int*)&ldsK[r * HEAD_DIM];
    unsigned int* v32 = (unsigned int*)&ldsV[r * HEAD_DIM];
    if (r < NTOK) {
      int hh = wh * WS + r / WS;             // shifted-grid coords
      int wc = wwin * WS + r % WS;
      int ho = (hh + SHIFT) % HH;            // undo roll(-SHIFT): original coords
      int wo = (wc + SHIFT) % WW;
      size_t t = ((size_t)bb * HH + ho) * WW + wo;
      const unsigned int* src = (const unsigned int*)(qkv + t * (3 * DIM));
      const unsigned int* sq = src + (head * HEAD_DIM) / 2;
      const unsigned int* sk = src + (DIM + head * HEAD_DIM) / 2;
      const unsigned int* sv = src + (2 * DIM + head * HEAD_DIM) / 2;
#pragma unroll
      for (int d = 0; d < HEAD_DIM / 2; ++d) { q32[d] = sq[d]; k32[d] = sk[d]; v32[d] = sv[d]; }
    } else {
#pragma unroll
      for (int d = 0; d < HEAD_DIM / 2; ++d) { q32[d] = 0u; k32[d] = 0u; v32[d] = 0u; }
    }
  }
  __syncthreads();

  // S = q k^T (q pre-scaled) + fused bias/mask table   -> ldsS (64x64 f32)
  for (int mt = 0; mt < 4; ++mt) {
    v16bf a = load_frag_rowmajor(ldsQ, mt * 16, HEAD_DIM, 0, lane);
    for (int nt = 0; nt < 4; ++nt) {
      v16bf bf = load_frag_rowmajor(ldsK, nt * 16, HEAD_DIM, 0, lane);
      v8f s = zero8();
      s = wmma_bf16(a, bf, s);
#pragma unroll
      for (int r = 0; r < 8; ++r) {
        int n = mt * 16 + r + ((lane >> 4) << 3);   // query token
        int m = nt * 16 + (lane & 15);              // key token
        ldsS[n * NPAD + m] = s[r] + tbase[n * NPAD + m];
      }
    }
  }
  __syncthreads();

  // softmax over keys (2 rows per lane) -> ldsP (bf16, overlays Q/K)
  for (int rr = 0; rr < 2; ++rr) {
    int rowi = lane + rr * 32;
    float* srow = &ldsS[rowi * NPAD];
    float mx = -1e30f;
    for (int j = 0; j < NPAD; ++j) mx = fmaxf(mx, srow[j]);
    float sum = 0.0f;
    for (int j = 0; j < NPAD; ++j) { float e = __expf(srow[j] - mx); srow[j] = e; sum += e; }
    float inv = 1.0f / sum;
    for (int j = 0; j < NPAD; ++j) ldsP[rowi * NPAD + j] = f32_to_bf16(srow[j] * inv);
  }
  __syncthreads();

  // out = P @ V, K=64 (two 32-steps)
  for (int mt = 0; mt < 4; ++mt) {
    for (int nt = 0; nt < 2; ++nt) {
      v8f o = zero8();
#pragma unroll
      for (int s = 0; s < 2; ++s) {
        v16bf a  = load_frag_rowmajor(ldsP, mt * 16, NPAD, s * 32, lane);
        v16bf bf = load_frag_colstrided(ldsV, nt * 16, HEAD_DIM, s * 32, lane);
        o = wmma_bf16(a, bf, o);
      }
#pragma unroll
      for (int r = 0; r < 8; ++r) {
        int n = mt * 16 + r + ((lane >> 4) << 3);
        int d = nt * 16 + (lane & 15);
        if (n < NTOK)
          attnout[((size_t)win * NTOK + n) * DIM + head * HEAD_DIM + d] = f32_to_bf16(o[r]);
      }
    }
  }
}

// ---------------- launcher ----------------
extern "C" void kernel_launch(void* const* d_in, const int* in_sizes, int n_in,
                              void* d_out, int out_size, void* d_ws, size_t ws_size,
                              hipStream_t stream) {
  (void)in_sizes; (void)n_in; (void)out_size; (void)ws_size;
  const float* x       = (const float*)d_in[0];
  const float* gamma1  = (const float*)d_in[1];
  const float* beta1   = (const float*)d_in[2];
  const float* w_qkv   = (const float*)d_in[3];
  const float* b_qkv   = (const float*)d_in[4];
  const float* rel_t   = (const float*)d_in[5];
  const float* w_proj  = (const float*)d_in[6];
  const float* b_proj  = (const float*)d_in[7];
  const float* gamma2  = (const float*)d_in[8];
  const float* beta2   = (const float*)d_in[9];
  const float* w_fc1   = (const float*)d_in[10];
  const float* b_fc1   = (const float*)d_in[11];
  const float* w_fc2   = (const float*)d_in[12];
  const float* b_fc2   = (const float*)d_in[13];
  float* out           = (float*)d_out;

  // workspace partition (256B aligned)
  char* w = (char*)d_ws;
  auto take = [&](size_t bytes) { void* p = (void*)w; w += (bytes + 255) & ~(size_t)255; return p; };
  unsigned short* lnbuf   = (unsigned short*)take((size_t)M_TOK * DIM * 2);
  unsigned short* wqkv_t  = (unsigned short*)take((size_t)DIM * 3 * DIM * 2);
  unsigned short* wproj_t = (unsigned short*)take((size_t)DIM * DIM * 2);
  unsigned short* wfc1_t  = (unsigned short*)take((size_t)DIM * MLP_HIDDEN * 2);
  unsigned short* wfc2_t  = (unsigned short*)take((size_t)MLP_HIDDEN * DIM * 2);
  float*          bm_tab  = (float*)take((size_t)NCLS * HEADS * NPAD * NPAD * 4);
  unsigned short* qkv_b   = (unsigned short*)take((size_t)M_TOK * 3 * DIM * 2);
  unsigned short* attno_b = (unsigned short*)take((size_t)M_TOK * DIM * 2);
  float*          projo_f = (float*)take((size_t)M_TOK * DIM * 4);
  float*          hbuf    = (float*)take((size_t)M_TOK * DIM * 4);
  unsigned short* ln2buf  = (unsigned short*)take((size_t)M_TOK * DIM * 2);
  unsigned short* fc1o_b  = (unsigned short*)take((size_t)M_TOK * MLP_HIDDEN * 2);

  // weights -> bf16, transposed to N-major (GEMM B operand)
  auto cvtT = [&](const float* src, unsigned short* dst, int K, int N) {
    int n = K * N;
    cvt_transpose_kernel<<<(n + 255) / 256, 256, 0, stream>>>(src, dst, K, N);
  };
  cvtT(w_qkv,  wqkv_t,  DIM, 3 * DIM);
  cvtT(w_proj, wproj_t, DIM, DIM);
  cvtT(w_fc1,  wfc1_t,  DIM, MLP_HIDDEN);
  cvtT(w_fc2,  wfc2_t,  MLP_HIDDEN, DIM);

  // fused bias+mask table
  {
    int n = NCLS * HEADS * NPAD * NPAD;
    build_biasmask_kernel<<<(n + 255) / 256, 256, 0, stream>>>(rel_t, bm_tab);
  }

  // 1) LN1 -> bf16 (original token order)
  ln1_kernel<<<M_TOK, 128, 0, stream>>>(x, gamma1, beta1, lnbuf);

  // 2) QKV GEMM: (100352 x 384) x (384 x 1152), q cols scaled
  gemm_wmma<0><<<dim3(3 * DIM / 128, M_TOK / 128), 256, 0, stream>>>(
      lnbuf, wqkv_t, b_qkv, nullptr, qkv_b, nullptr, M_TOK, DIM, 3 * DIM);

  // 3) windowed shifted attention (one wave per window-head)
  attn_kernel<<<BATCH * NWIN * HEADS, 32, 0, stream>>>(qkv_b, bm_tab, attno_b);

  // 4) proj GEMM (windowed rows), f32 out
  gemm_wmma<1><<<dim3(DIM / 128, M_TOK / 128), 256, 0, stream>>>(
      attno_b, wproj_t, b_proj, projo_f, nullptr, nullptr, M_TOK, DIM, DIM);

  // 5) window-reverse + residual + LN2
  residual_ln2_kernel<<<M_TOK, 128, 0, stream>>>(x, projo_f, gamma2, beta2, hbuf, ln2buf);

  // 6) fc1 GEMM + GELU -> bf16
  gemm_wmma<2><<<dim3(MLP_HIDDEN / 128, M_TOK / 128), 256, 0, stream>>>(
      ln2buf, wfc1_t, b_fc1, nullptr, fc1o_b, nullptr, M_TOK, DIM, MLP_HIDDEN);

  // 7) fc2 GEMM + bias + residual -> d_out (f32)
  gemm_wmma<3><<<dim3(DIM / 128, M_TOK / 128), 256, 0, stream>>>(
      fc1o_b, wfc2_t, b_fc2, out, nullptr, hbuf, M_TOK, MLP_HIDDEN, DIM);
}
